// GNN_66907000537683
// MI455X (gfx1250) — compile-verified, matrix operand order checked by reading, not run on hardware
//
#include <hip/hip_runtime.h>
#include <hip/hip_bf16.h>

#define NNODES   100000
#define NEDGES   1600000
#define DIM      64
#define NLAYERS  3
#define BN_EPS   1e-5f
#define AGG_EPS  1e-6f

typedef __attribute__((ext_vector_type(2)))  float    v2f;
typedef __attribute__((ext_vector_type(8)))  float    v8f;
typedef __attribute__((ext_vector_type(16))) _Float16 v16h;

// ---------------------------------------------------------------------------
// GEMM: out = h @ W^T + b for 4 weight matrices at once.
// Block = 256 threads = 8 waves, handles 16 node rows.
// wave -> (matrix = wave/2, col tiles {0,1} or {2,3}).
// ---------------------------------------------------------------------------
__global__ __launch_bounds__(256) void ggcn_gemm4(
    const float* __restrict__ h,
    const float* __restrict__ Wa, const float* __restrict__ ba,
    const float* __restrict__ Wb, const float* __restrict__ bb,
    const float* __restrict__ Wd, const float* __restrict__ bd,
    const float* __restrict__ We, const float* __restrict__ be,
    float* __restrict__ Oa, float* __restrict__ Ob,
    float* __restrict__ Od, float* __restrict__ Oe)
{
    const int wave = threadIdx.x >> 5;
    const int lane = threadIdx.x & 31;
    const int hi   = lane >> 4;     // 0/1: K/M sub-half per ISA layout
    const int mn   = lane & 15;     // M for A-tile, N for B/C/D tiles
    const int rowBase = blockIdx.x * 16;
    if (rowBase >= NNODES) return;

    const int mat  = wave >> 1;
    const int col0 = (wave & 1) * 32;       // first of two 16-wide col tiles
    const int col1 = col0 + 16;

    const float* W; const float* bias; float* O;
    switch (mat) {
        case 0:  W = Wa; bias = ba; O = Oa; break;
        case 1:  W = Wb; bias = bb; O = Ob; break;
        case 2:  W = Wd; bias = bd; O = Od; break;
        default: W = We; bias = be; O = Oe; break;
    }

    const float* hrow  = h + (size_t)(rowBase + mn) * DIM;
    const float* wrow0 = W + (size_t)(col0 + mn) * DIM;
    const float* wrow1 = W + (size_t)(col1 + mn) * DIM;

    v8f c0 = {}; v8f c1 = {};

#if defined(__AMDGCN__) && __has_builtin(__builtin_amdgcn_wmma_f32_16x16x4_f32)
    // fp32 WMMA, K=4 per step, 16 steps for K=64.
    // A 16x4 f32: lane holds M=mn, VGPR j = K = k0 + 2*hi + j  -> float2 load.
    #pragma unroll
    for (int kk = 0; kk < 16; ++kk) {
        const int k0 = kk * 4 + hi * 2;
        v2f a  = *(const v2f*)(hrow  + k0);
        v2f b0 = *(const v2f*)(wrow0 + k0);
        v2f b1 = *(const v2f*)(wrow1 + k0);
        c0 = __builtin_amdgcn_wmma_f32_16x16x4_f32(false, a, false, b0,
                                                   (short)0, c0, false, false);
        c1 = __builtin_amdgcn_wmma_f32_16x16x4_f32(false, a, false, b1,
                                                   (short)0, c1, false, false);
    }
#else
    // Fallback: f16 WMMA 16x16x32, two K-blocks. 16-bit A layout:
    // elems 0..7 -> K = kb + 8*hi + j ; elems 8..15 -> K = kb + 16 + 8*hi + j
    for (int kb = 0; kb < 64; kb += 32) {
        v16h a, b0, b1;
        #pragma unroll
        for (int j = 0; j < 8; ++j) {
            a [j]     = (_Float16)hrow [kb + 8 * hi + j];
            a [j + 8] = (_Float16)hrow [kb + 16 + 8 * hi + j];
            b0[j]     = (_Float16)wrow0[kb + 8 * hi + j];
            b0[j + 8] = (_Float16)wrow0[kb + 16 + 8 * hi + j];
            b1[j]     = (_Float16)wrow1[kb + 8 * hi + j];
            b1[j + 8] = (_Float16)wrow1[kb + 16 + 8 * hi + j];
        }
        c0 = __builtin_amdgcn_wmma_f32_16x16x32_f16(false, a, false, b0,
                                                    (short)0, c0, false, false);
        c1 = __builtin_amdgcn_wmma_f32_16x16x32_f16(false, a, false, b1,
                                                    (short)0, c1, false, false);
    }
#endif

    const float bias0 = bias[col0 + mn];
    const float bias1 = bias[col1 + mn];
    #pragma unroll
    for (int v = 0; v < 8; ++v) {
        const size_t row = (size_t)(rowBase + v + 8 * hi);
        O[row * DIM + col0 + mn] = c0[v] + bias0;
        O[row * DIM + col1 + mn] = c1[v] + bias1;
    }
}

// ---------------------------------------------------------------------------
// Ce[d] = sum_k WC[d][k] + bC[d]   (one 64-thread block)
// ---------------------------------------------------------------------------
__global__ void ggcn_ce(const float* __restrict__ WC,
                        const float* __restrict__ bC,
                        float* __restrict__ Ce)
{
    const int d = threadIdx.x;
    if (d >= DIM) return;
    float s = bC[d];
    #pragma unroll 8
    for (int k = 0; k < DIM; ++k) s += WC[d * DIM + k];
    Ce[d] = s;
}

// ---------------------------------------------------------------------------
// Zero scratch region (agg + cnt + stat sums)
// ---------------------------------------------------------------------------
__global__ void ggcn_zero(float* __restrict__ p, size_t n)
{
    size_t i = (size_t)blockIdx.x * blockDim.x + threadIdx.x;
    const size_t stride = (size_t)gridDim.x * blockDim.x;
    for (; i < n; i += stride) p[i] = 0.0f;
}

// ---------------------------------------------------------------------------
// Edge kernel: one wave per edge, each lane owns 2 features.
//   sigma = sigmoid(Dh[row] + Eh[col] + Ce); msg = sigma * Bh[row]
//   agg[col] += msg (fp32 atomics, L2-resident); cnt[col] += 1
// ---------------------------------------------------------------------------
__global__ __launch_bounds__(256) void ggcn_edge(
    const int* __restrict__ ei,            // [2, E] int32
    const float* __restrict__ Bh, const float* __restrict__ Dh,
    const float* __restrict__ Eh, const float* __restrict__ Ce,
    float* __restrict__ agg, float* __restrict__ cnt)
{
    const int wave = threadIdx.x >> 5;
    const int lane = threadIdx.x & 31;
    const long long e = (long long)blockIdx.x * 8 + wave;
    if (e >= NEDGES) return;

    const int row = ei[e];                 // j (source)
    const int col = ei[(size_t)NEDGES + e];// i (target)
    const int f   = lane * 2;
    const size_t ro = (size_t)row * DIM + f;
    const size_t co = (size_t)col * DIM + f;

    v2f d  = *(const v2f*)(Dh + ro);
    v2f ee = *(const v2f*)(Eh + co);
    v2f cc = *(const v2f*)(Ce + f);
    v2f b  = *(const v2f*)(Bh + ro);

    const float gx = d.x + ee.x + cc.x;
    const float gy = d.y + ee.y + cc.y;
    const float sx = 1.0f / (1.0f + __expf(-gx));
    const float sy = 1.0f / (1.0f + __expf(-gy));

    atomicAdd(&agg[co],     sx * b.x);
    atomicAdd(&agg[co + 1], sy * b.y);
    if (lane == 0) atomicAdd(&cnt[col], 1.0f);
}

// ---------------------------------------------------------------------------
// h_pre = Ah + agg/(cnt+eps) (in-place into Ah) + per-column sum/sumsq partials
// ---------------------------------------------------------------------------
__global__ __launch_bounds__(256) void ggcn_node_stats(
    float* __restrict__ hpre, const float* __restrict__ agg,
    const float* __restrict__ cnt, float* __restrict__ sums /* [128] */)
{
    __shared__ float ls[512];
    const int d = threadIdx.x & 63;
    const int g = threadIdx.x >> 6;        // 0..3 node sub-group
    float s = 0.0f, ss = 0.0f;
    for (int n = blockIdx.x * 4 + g; n < NNODES; n += gridDim.x * 4) {
        const size_t off = (size_t)n * DIM + d;
        const float inv = 1.0f / (cnt[n] + AGG_EPS);
        const float v = hpre[off] + agg[off] * inv;
        hpre[off] = v;
        s += v; ss += v * v;
    }
    ls[threadIdx.x]       = s;
    ls[256 + threadIdx.x] = ss;
    __syncthreads();
    if (g == 0) {
        float ts = 0.0f, tss = 0.0f;
        #pragma unroll
        for (int j = 0; j < 4; ++j) {
            ts  += ls[j * 64 + d];
            tss += ls[256 + j * 64 + d];
        }
        atomicAdd(&sums[d],      ts);
        atomicAdd(&sums[64 + d], tss);
    }
}

// ---------------------------------------------------------------------------
// BatchNorm (batch stats, biased var) + affine + ReLU -> h_out
// ---------------------------------------------------------------------------
__global__ __launch_bounds__(256) void ggcn_bn_relu(
    const float* __restrict__ hpre, const float* __restrict__ sums,
    const float* __restrict__ gamma, const float* __restrict__ beta,
    float* __restrict__ hout)
{
    const size_t i = (size_t)blockIdx.x * blockDim.x + threadIdx.x;
    if (i >= (size_t)NNODES * DIM) return;
    const int d = (int)(i & 63);
    const float invN = 1.0f / (float)NNODES;
    const float mu  = sums[d] * invN;
    const float var = sums[64 + d] * invN - mu * mu;
    const float v = (hpre[i] - mu) * rsqrtf(var + BN_EPS) * gamma[d] + beta[d];
    hout[i] = v > 0.0f ? v : 0.0f;
}

// ---------------------------------------------------------------------------
extern "C" void kernel_launch(void* const* d_in, const int* in_sizes, int n_in,
                              void* d_out, int out_size, void* d_ws, size_t ws_size,
                              hipStream_t stream) {
    (void)in_sizes; (void)n_in; (void)out_size; (void)ws_size;

    const float* x     = (const float*)d_in[0];
    const int*   ei    = (const int*)  d_in[1];
    const float* WA    = (const float*)d_in[2];
    const float* bA    = (const float*)d_in[3];
    const float* WB    = (const float*)d_in[4];
    const float* bB    = (const float*)d_in[5];
    const float* WC    = (const float*)d_in[6];
    const float* bC    = (const float*)d_in[7];
    const float* WD    = (const float*)d_in[8];
    const float* bD    = (const float*)d_in[9];
    const float* WE    = (const float*)d_in[10];
    const float* bE    = (const float*)d_in[11];
    const float* gamma = (const float*)d_in[12];
    const float* beta  = (const float*)d_in[13];
    float* out = (float*)d_out;

    const size_t ND = (size_t)NNODES * DIM;
    float* ws   = (float*)d_ws;
    float* Ah   = ws;                  // [N,64] -> becomes h_pre
    float* Bh   = Ah  + ND;            // [N,64]
    float* Dh   = Bh  + ND;            // [N,64]
    float* Eh   = Dh  + ND;            // [N,64]
    float* agg  = Eh  + ND;            // [N,64]   zeroed per layer
    float* cnt  = agg + ND;            // [N]      zeroed per layer
    float* sums = cnt + NNODES;        // [128]    zeroed per layer
    float* Ce   = sums + 128;          // [64]

    const size_t zeroCount = ND + NNODES + 128;   // agg + cnt + sums
    const int gemmBlocks   = (NNODES + 15) / 16;
    const int edgeBlocks   = (NEDGES + 7) / 8;
    const int elemBlocks   = (int)((ND + 255) / 256);

    for (int l = 0; l < NLAYERS; ++l) {
        const size_t wOff = (size_t)l * DIM * DIM;
        const size_t vOff = (size_t)l * DIM;
        const float* h_in = (l == 0) ? x : out;

        ggcn_zero<<<2048, 256, 0, stream>>>(agg, zeroCount);
        ggcn_ce<<<1, 64, 0, stream>>>(WC + wOff, bC + vOff, Ce);
        ggcn_gemm4<<<gemmBlocks, 256, 0, stream>>>(
            h_in,
            WA + wOff, bA + vOff, WB + wOff, bB + vOff,
            WD + wOff, bD + vOff, WE + wOff, bE + vOff,
            Ah, Bh, Dh, Eh);
        ggcn_edge<<<edgeBlocks, 256, 0, stream>>>(ei, Bh, Dh, Eh, Ce, agg, cnt);
        ggcn_node_stats<<<4096, 256, 0, stream>>>(Ah, agg, cnt, sums);
        ggcn_bn_relu<<<elemBlocks, 256, 0, stream>>>(
            Ah, sums, gamma + vOff, beta + vOff, out);
    }
}
